// CausalSelfAttention_8615704396103
// MI455X (gfx1250) — matrix operand order repaired and checked by reading, not compiled, not run on hardware
//
#include <hip/hip_runtime.h>
#include <stdint.h>

// Problem constants (match reference)
#define B_ 2
#define T_ 2048
#define C_ 1024
#define H_ 16
#define D_ 64

typedef __bf16 bf16_t;
typedef __attribute__((ext_vector_type(16))) __bf16 v16bf;
typedef __attribute__((ext_vector_type(8)))  __bf16 v8bf;
typedef __attribute__((ext_vector_type(8)))  float  v8f;

static __device__ __forceinline__ bf16_t f2bf(float f) {
  union { float f; uint32_t u; } v; v.f = f;
  uint32_t r = (v.u + 0x7FFFu + ((v.u >> 16) & 1u)) >> 16;  // RNE
  union { unsigned short s; bf16_t b; } o; o.s = (unsigned short)r;
  return o.b;
}

// ---------------------------------------------------------------------------
// Kernel 1: fp32 -> bf16 cast (grid-stride)
// ---------------------------------------------------------------------------
__global__ __launch_bounds__(256) void cast_f32_bf16(const float* __restrict__ in,
                                                     bf16_t* __restrict__ out, int n) {
  for (int i = blockIdx.x * blockDim.x + threadIdx.x; i < n;
       i += gridDim.x * blockDim.x)
    out[i] = f2bf(in[i]);
}

// ---------------------------------------------------------------------------
// Kernel 2: bf16 GEMM with WMMA.  OUT[M,N] = A[M,K] @ W[K,N] + bias
// Block = 256 threads (8 waves).  Tile: 64(M) x 128(N), K-step 32.
// Wave (w): waveM = w&3 (16 rows), waveN = w>>2 (64 cols -> 4 WMMA tiles).
// One A fragment is reused across 4 B tiles (4 WMMAs between barriers).
// WRITE_F32: 1 -> write float output, 0 -> write bf16 output
// ---------------------------------------------------------------------------
template <int WRITE_F32>
__global__ __launch_bounds__(256) void gemm_bf16_wmma(
    const bf16_t* __restrict__ A, const bf16_t* __restrict__ W,
    const float* __restrict__ bias, bf16_t* __restrict__ outb,
    float* __restrict__ outf, int M, int N, int K) {
  __shared__ bf16_t lA[64 * 32];    // A tile, row-major [m][k]    (4 KB)
  __shared__ bf16_t lWt[128 * 32];  // W tile transposed [n][k]    (8 KB)

  const int tid  = threadIdx.x;
  const int lane = tid & 31;
  const int wave = tid >> 5;
  const int waveM = wave & 3;   // 16-row slice
  const int waveN = wave >> 2;  // 64-col slice
  const int bM = blockIdx.y * 64;
  const int bN = blockIdx.x * 128;

  v8f acc[4] = {v8f{}, v8f{}, v8f{}, v8f{}};

  for (int k0 = 0; k0 < K; k0 += 32) {
    // --- stage A tile: 64x32, one 16B vector load per thread
    {
      int row = tid >> 2;
      int cc  = (tid & 3) * 8;
      *(v8bf*)&lA[row * 32 + cc] =
          *(const v8bf*)(A + (size_t)(bM + row) * K + k0 + cc);
    }
    // --- stage W tile transposed: global [k][n] -> LDS [n][k], 16 elems/thr
    {
      int k  = tid & 31;
      int nc = (tid >> 5) * 16;
      const bf16_t* src = W + (size_t)(k0 + k) * N + bN + nc;
      v8bf w0 = *(const v8bf*)(src);
      v8bf w1 = *(const v8bf*)(src + 8);
#pragma unroll
      for (int j = 0; j < 8; ++j) {
        lWt[(nc + j) * 32 + k]     = w0[j];
        lWt[(nc + 8 + j) * 32 + k] = w1[j];
      }
    }
    __syncthreads();

    // --- A fragment (16x32): lane holds row m=lane%16; lanes0-15: K 0-7 &
    //     16-23, lanes16-31: K 8-15 & 24-31  -> two contiguous 16B LDS loads
    const int mrow = waveM * 16 + (lane & 15);
    const int kbA  = (lane >> 4) * 8;
    union { v16bf v; v8bf h[2]; } ua;
    ua.h[0] = *(const v8bf*)&lA[mrow * 32 + kbA];
    ua.h[1] = *(const v8bf*)&lA[mrow * 32 + kbA + 16];

    // --- B fragments (32x16): lane holds col n; lanes0-15: K 0-7 & 16-23,
    //     lanes16-31: K 8-15 & 24-31 -> two 16B loads from transposed tile
    const int kbB = (lane >> 4) * 8;
#pragma unroll
    for (int t = 0; t < 4; ++t) {
      int ncol = waveN * 64 + t * 16 + (lane & 15);
      union { v16bf v; v8bf h[2]; } ub;
      ub.h[0] = *(const v8bf*)&lWt[ncol * 32 + kbB];
      ub.h[1] = *(const v8bf*)&lWt[ncol * 32 + kbB + 16];
      acc[t] = __builtin_amdgcn_wmma_f32_16x16x32_bf16(
          false, ua.v, false, ub.v, (short)0, acc[t], false, false);
    }
    __syncthreads();
  }

  // --- epilogue: C/D layout: lane -> (M = r + 8*(lane>=16), N = lane%16)
#pragma unroll
  for (int t = 0; t < 4; ++t) {
    int col = bN + waveN * 64 + t * 16 + (lane & 15);
    float bv = bias[col];
#pragma unroll
    for (int r = 0; r < 8; ++r) {
      int row = bM + waveM * 16 + (lane >> 4) * 8 + r;
      float val = acc[t][r] + bv;
      if (WRITE_F32)
        outf[(size_t)row * N + col] = val;
      else
        outb[(size_t)row * N + col] = f2bf(val);
    }
  }
}

// ---------------------------------------------------------------------------
// Kernel 3: flash attention, one wave per (b, h, 16-row Q tile).
// 32 keys per iteration: S uses 4 WMMAs (2 key-tiles x K=64), P.V uses 4
// fully-packed K=32 WMMAs (no zero padding, all lanes live).
// qkv: bf16 [B*T, 3C] row-major; q|k|v at col offsets 0|C|2C, head h at h*D.
// y:   bf16 [B*T, C]
// ---------------------------------------------------------------------------
__global__ __launch_bounds__(32) void attn_flash(const bf16_t* __restrict__ qkv,
                                                 bf16_t* __restrict__ y) {
  __shared__ bf16_t lP[16 * 32];   // P tile (bf16) [m][key]   (1 KB)
  __shared__ bf16_t lVt[64 * 32];  // V tile transposed [d][key] (4 KB)

  const int lane = threadIdx.x;
  const int qt = blockIdx.x;  // q tile (16 rows)
  const int h  = blockIdx.y;
  const int b  = blockIdx.z;
  const int ldq = 3 * C_;
  const int q0 = qt * 16;

  const bf16_t* qbase = qkv + (size_t)b * T_ * ldq + h * D_;
  const bf16_t* kbase = qbase + C_;
  const bf16_t* vbase = qbase + 2 * C_;

  const int n = lane & 15;            // C-layout column / B-operand column
  const int rbase = (lane >> 4) * 8;  // C-layout row base for this lane half

  // Q fragments: A operand, two 16x32 slices over D=64
  const int qrow = q0 + n;  // A lane row = lane%16
  v16bf aq[2];
#pragma unroll
  for (int j = 0; j < 2; ++j) {
    int kb = j * 32 + (lane >> 4) * 8;
    union { v16bf v; v8bf h2[2]; } u;
    u.h2[0] = *(const v8bf*)(qbase + (size_t)qrow * ldq + kb);
    u.h2[1] = *(const v8bf*)(qbase + (size_t)qrow * ldq + kb + 16);
    aq[j] = u.v;
  }

  v8f acc[4] = {v8f{}, v8f{}, v8f{}, v8f{}};
  float mrun[8], lrun[8];
#pragma unroll
  for (int r = 0; r < 8; ++r) { mrun[r] = -3.0e38f; lrun[r] = 0.0f; }

  const float scale = 0.125f;  // 1/sqrt(64)

  // key blocks of 32; last block may cross the causal diagonal
  for (int kb = 0; kb < q0 + 16; kb += 32) {
    __syncthreads();  // previous iteration's LDS reads done before overwrite

    // ---- S = Q @ K^T, two 16-key column tiles (kt2 = 0,1)
    v8f s[2] = {v8f{}, v8f{}};
#pragma unroll
    for (int kt2 = 0; kt2 < 2; ++kt2) {
      const bf16_t* krow = kbase + (size_t)(kb + kt2 * 16 + n) * ldq;
#pragma unroll
      for (int j = 0; j < 2; ++j) {
        int kk = j * 32 + (lane >> 4) * 8;
        union { v16bf v; v8bf h2[2]; } u;
        u.h2[0] = *(const v8bf*)(krow + kk);
        u.h2[1] = *(const v8bf*)(krow + kk + 16);
        s[kt2] = __builtin_amdgcn_wmma_f32_16x16x32_bf16(
            false, aq[j], false, u.v, (short)0, s[kt2], false, false);
      }
    }

    // ---- stage V tile transposed into LDS: Vt[d][key], key = lane
    {
      const bf16_t* src = vbase + (size_t)(kb + lane) * ldq;
#pragma unroll
      for (int c = 0; c < 8; ++c) {
        v8bf vv = *(const v8bf*)(src + c * 8);
#pragma unroll
        for (int j = 0; j < 8; ++j) lVt[(c * 8 + j) * 32 + lane] = vv[j];
      }
    }

    // ---- online softmax over 32 keys per row
    const bool needMask = (kb + 31 > q0);
#pragma unroll
    for (int r = 0; r < 8; ++r) {
      float v0 = s[0][r] * scale;
      float v1 = s[1][r] * scale;
      if (needMask) {
        int rowAbs = q0 + rbase + r;             // absolute q position
        if (kb + n > rowAbs)      v0 = -3.0e38f; // key index kb+n
        if (kb + 16 + n > rowAbs) v1 = -3.0e38f; // key index kb+16+n
      }
      float mx = fmaxf(v0, v1);
      mx = fmaxf(mx, __shfl_xor(mx, 1, 32));
      mx = fmaxf(mx, __shfl_xor(mx, 2, 32));
      mx = fmaxf(mx, __shfl_xor(mx, 4, 32));
      mx = fmaxf(mx, __shfl_xor(mx, 8, 32));
      float mnew = fmaxf(mrun[r], mx);
      float p0 = __expf(v0 - mnew);
      float p1 = __expf(v1 - mnew);
      float rs = p0 + p1;
      rs += __shfl_xor(rs, 1, 32);
      rs += __shfl_xor(rs, 2, 32);
      rs += __shfl_xor(rs, 4, 32);
      rs += __shfl_xor(rs, 8, 32);
      float corr = __expf(mrun[r] - mnew);
      lrun[r] = lrun[r] * corr + rs;
      mrun[r] = mnew;
#pragma unroll
      for (int t = 0; t < 4; ++t) acc[t][r] *= corr;
      lP[(rbase + r) * 32 + n]      = f2bf(p0);  // keys 0..15 of block
      lP[(rbase + r) * 32 + 16 + n] = f2bf(p1);  // keys 16..31 of block
    }
    __syncthreads();

    // ---- O += P @ V : fully packed 16x16x32 WMMAs (K = 32 keys)
    union { v16bf v; v8bf h2[2]; } uap;
    {
      int kk = (lane >> 4) * 8;  // A layout: lanes0-15 K0-7/16-23, 16-31: 8-15/24-31
      uap.h2[0] = *(const v8bf*)&lP[n * 32 + kk];
      uap.h2[1] = *(const v8bf*)&lP[n * 32 + kk + 16];
    }
    const int kkB = (lane >> 4) * 8;  // B layout: lanes0-15 K0-7/16-23, 16-31: 8-15/24-31
#pragma unroll
    for (int t = 0; t < 4; ++t) {
      int d = t * 16 + n;
      union { v16bf v; v8bf h2[2]; } ub;
      ub.h2[0] = *(const v8bf*)&lVt[d * 32 + kkB];
      ub.h2[1] = *(const v8bf*)&lVt[d * 32 + kkB + 16];
      acc[t] = __builtin_amdgcn_wmma_f32_16x16x32_bf16(
          false, uap.v, false, ub.v, (short)0, acc[t], false, false);
    }
  }

  // ---- normalize and write y (bf16)
#pragma unroll
  for (int t = 0; t < 4; ++t) {
#pragma unroll
    for (int r = 0; r < 8; ++r) {
      int row = q0 + rbase + r;
      int d = t * 16 + n;
      float val = acc[t][r] / lrun[r];
      y[(size_t)(b * T_ + row) * C_ + h * D_ + d] = f2bf(val);
    }
  }
}

// ---------------------------------------------------------------------------
// Launcher
// ---------------------------------------------------------------------------
extern "C" void kernel_launch(void* const* d_in, const int* in_sizes, int n_in,
                              void* d_out, int out_size, void* d_ws,
                              size_t ws_size, hipStream_t stream) {
  const float* x      = (const float*)d_in[0];
  const float* w_qkv  = (const float*)d_in[1];
  const float* b_qkv  = (const float*)d_in[2];
  const float* w_proj = (const float*)d_in[3];
  const float* b_proj = (const float*)d_in[4];
  float* out = (float*)d_out;

  char* ws = (char*)d_ws;
  size_t o = 0;
  auto alloc = [&](size_t bytes) -> void* {
    void* p = ws + o;
    o += (bytes + 255) & ~(size_t)255;
    return p;
  };
  const size_t nX = (size_t)B_ * T_ * C_;        // 4,194,304
  const size_t nWq = (size_t)C_ * 3 * C_;        // 3,145,728
  const size_t nWp = (size_t)C_ * C_;            // 1,048,576
  const size_t nQKV = (size_t)B_ * T_ * 3 * C_;  // 12,582,912

  bf16_t* x_bf     = (bf16_t*)alloc(nX * 2);
  bf16_t* wqkv_bf  = (bf16_t*)alloc(nWq * 2);
  bf16_t* wproj_bf = (bf16_t*)alloc(nWp * 2);
  bf16_t* qkv_bf   = (bf16_t*)alloc(nQKV * 2);
  bf16_t* y_bf     = (bf16_t*)alloc(nX * 2);

  // 1) casts
  cast_f32_bf16<<<2048, 256, 0, stream>>>(x, x_bf, (int)nX);
  cast_f32_bf16<<<2048, 256, 0, stream>>>(w_qkv, wqkv_bf, (int)nWq);
  cast_f32_bf16<<<2048, 256, 0, stream>>>(w_proj, wproj_bf, (int)nWp);

  // 2) QKV projection: [4096,1024] @ [1024,3072] + b_qkv -> bf16
  {
    dim3 grid(3 * C_ / 128, B_ * T_ / 64);
    gemm_bf16_wmma<0><<<grid, 256, 0, stream>>>(
        x_bf, wqkv_bf, b_qkv, qkv_bf, nullptr, B_ * T_, 3 * C_, C_);
  }

  // 3) causal flash attention -> y_bf
  {
    dim3 grid(T_ / 16, H_, B_);
    attn_flash<<<grid, 32, 0, stream>>>(qkv_bf, y_bf);
  }

  // 4) output projection: [4096,1024] @ [1024,1024] + b_proj -> fp32 d_out
  {
    dim3 grid(C_ / 128, B_ * T_ / 64);
    gemm_bf16_wmma<1><<<grid, 256, 0, stream>>>(
        y_bf, wproj_bf, b_proj, nullptr, out, B_ * T_, C_, C_);
  }
}
